// PixelCNNDistribution_34574486733013
// MI455X (gfx1250) — matrix operand order repaired and checked by reading, not compiled, not run on hardware
//
#include <hip/hip_runtime.h>
#include <hip/hip_bf16.h>

// ---------------------------------------------------------------------------
// PixelCNN forward on gfx1250: f16 WMMA implicit-GEMM.
//  - feature maps stored POST-ReLU, f16 channels-last -> load paths are pure
//    copies (async global->LDS), ReLU fused into the single store site.
//  - each wave computes a 16(M) x 32(N) tile: A-fragments (weights) feed two
//    WMMAs each, halving global weight traffic.
// ---------------------------------------------------------------------------

typedef __attribute__((ext_vector_type(16))) _Float16 v16h;
typedef __attribute__((ext_vector_type(8)))  _Float16 v8h;
typedef __attribute__((ext_vector_type(8)))  float    v8f;

union Frag16 { v16h v; v8h h[2]; };

#define HID   128
#define CCH   64
#define NPIX  1024          // 32*32 per image
#define HALOC 34            // 32 + 2 halo columns

static __device__ __forceinline__ v8f wmma_f16(const Frag16& a, const Frag16& b,
                                               v8f c) {
    return __builtin_amdgcn_wmma_f32_16x16x32_f16(false, a.v, false, b.v,
                                                  (short)0, c, false, false);
}

// ---------------------------------------------------------------------------
// Pack: masked Wl -> f16 [6][128][5tap][128], Wc -> f16 [6][128][64],
// w_out -> f16 [768][128], h -> f16 channels-last [32][32][32][64]
// ---------------------------------------------------------------------------
__global__ void pixelcnn_pack_kernel(const float* __restrict__ Wl,
                                     const float* __restrict__ Wc,
                                     const float* __restrict__ wout,
                                     const float* __restrict__ h,
                                     _Float16* __restrict__ WlF,
                                     _Float16* __restrict__ WcF,
                                     _Float16* __restrict__ WoF,
                                     _Float16* __restrict__ hF)
{
    int i = blockIdx.x * 256 + threadIdx.x;

    if (i < 6 * HID * 5 * HID) {              // masked 3x3 weights, 5 live taps
        int c    = i & 127;
        int rest = i >> 7;                    // (l*128+o)*5 + t
        int t    = rest % 5;
        int lo   = rest / 5;                  // l*128+o
        int o    = lo & 127;
        int l    = lo >> 7;
        int ky   = (t < 3) ? 0 : 1;
        int kx   = (t < 3) ? t : (t - 3);
        WlF[i] = (_Float16)Wl[(((l * HID + o) * HID + c) * 3 + ky) * 3 + kx];
    }
    if (i < 6 * HID * CCH)  WcF[i] = (_Float16)Wc[i];     // same [l][o][c] order
    if (i < 768 * HID)      WoF[i] = (_Float16)wout[i];   // [o][c]
    if (i < 32 * NPIX * CCH) {                            // h: NCHW -> NHWC f16
        int c  = i & 63;
        int p  = i >> 6;                                  // global pixel
        int b  = p >> 10;
        int pp = p & 1023;
        hF[i] = (_Float16)h[(b * CCH + c) * NPIX + pp];
    }
}

// ---------------------------------------------------------------------------
// First conv: masked 7x7 type-A, 3 in-ch -> 128, scalar (tiny FLOP count).
// Writes POST-ReLU f16 channels-last [32768][128].
// ---------------------------------------------------------------------------
__global__ __launch_bounds__(256) void pixelcnn_conv0_kernel(
    const float* __restrict__ x,      // [32][3][32][32]
    const float* __restrict__ w0,     // [128][3][7][7]
    const float* __restrict__ b0,     // [128]
    _Float16* __restrict__ feat)      // [32768][128]
{
    int tid = threadIdx.x;
    int o   = tid & 127;
    int pi  = blockIdx.x * 2 + (tid >> 7);
    int b   = pi >> 10;
    int p   = pi & 1023;
    int y   = p >> 5;
    int xx  = p & 31;

    float acc = b0[o];
    // mask A: rows ky=0..2 full, ky==3 only kx<3 (center excluded), ky>3 zero
    for (int ky = 0; ky < 4; ++ky) {
        int kxmax = (ky < 3) ? 7 : 3;
        int gy = y + ky - 3;
        if (gy < 0) continue;
        for (int kx = 0; kx < kxmax; ++kx) {
            int gx = xx + kx - 3;
            if (gx < 0 || gx > 31) continue;
            #pragma unroll
            for (int ci = 0; ci < 3; ++ci)
                acc += x[((b * 3 + ci) * 32 + gy) * 32 + gx] *
                       w0[((o * 3 + ci) * 7 + ky) * 7 + kx];
        }
    }
    feat[pi * HID + o] = (_Float16)(acc > 0.f ? acc : 0.f);   // ReLU fused
}

// ---------------------------------------------------------------------------
// Layer: act_out = relu( maskedConv3x3(act_in) + Wc@h + Bl )
// Workgroup = 8 waves = one full 32-pixel row; wave m owns out-ch [16m,16m+16)
// across both 16-px N-subtiles (A-fragment reuse). Halo fill is a pure copy
// via GLOBAL_LOAD_ASYNC_TO_LDS_B128 (ASYNCcnt), borders zero via ds_store.
// 44 v_wmma per wave per row.
// ---------------------------------------------------------------------------
__global__ __launch_bounds__(256, 2) void pixelcnn_layer_kernel(
    const _Float16* __restrict__ actin,    // [32768][128] post-relu
    _Float16* __restrict__ actout,         // [32768][128] post-relu
    const _Float16* __restrict__ WlF,      // [6][128][5][128]
    const _Float16* __restrict__ WcF,      // [6][128][64]
    const _Float16* __restrict__ hF,       // [32768][64]
    const float* __restrict__ Bl,          // [6][128]
    int layer)
{
    __shared__ __align__(16) _Float16 s_feat[3 * HALOC * HID];  // 26112 B halo
    __shared__ __align__(16) _Float16 s_out[32 * HID];          // 8192 B

    const int tid = threadIdx.x;
    const int blk = blockIdx.x;
    const int y   = blk & 31;
    const int b   = blk >> 5;

    // ---- halo fill: 102 px * 16 chunks of 8ch (16B), async global->LDS ----
    for (int idx = tid; idx < 3 * HALOC * 16; idx += 256) {
        int pix = idx >> 4;                  // 0..101
        int c8  = idx & 15;
        int r   = pix / HALOC;
        int col = pix % HALOC;
        int gy  = y + r - 1;
        int gx  = col - 1;
        _Float16* lp = s_feat + pix * HID + c8 * 8;
        if (gy >= 0 && gy < 32 && gx >= 0 && gx < 32) {
            unsigned long long ga = (unsigned long long)
                (actin + (((b * 32 + gy) * 32 + gx) * HID + c8 * 8));
            unsigned lo = (unsigned)(unsigned long long)lp;   // LDS byte offset
            asm volatile("global_load_async_to_lds_b128 %0, %1, off"
                         :: "v"(lo), "v"(ga) : "memory");
        } else {
            v8h z = {};
            *(v8h*)lp = z;                   // zero-pad border (ds_store_b128)
        }
    }
    asm volatile("s_wait_asynccnt 0x0" ::: "memory");
    __syncthreads();

    const int wave = tid >> 5;        // M-tile: out channels [wave*16, +16)
    const int lane = tid & 31;
    const int lr   = lane & 15;       // A: row M; B/D: col N (pixel)
    const int lh   = lane >> 4;       // lane half -> contiguous K half [16*lh, +16)

    v8f acc0 = {};                    // pixels lr
    v8f acc1 = {};                    // pixels 16+lr

    // ---- 1x1 conditioning: Wc[l] (128x64) @ h (64x32), 2 K-chunks ----
    const _Float16* wcb = WcF + (layer * HID + wave * 16 + lr) * CCH;
    const _Float16* hb0 = hF + (((b * 32 + y) * 32 + lr) * CCH);
    const _Float16* hb1 = hb0 + 16 * CCH;
    #pragma unroll
    for (int c0 = 0; c0 < CCH; c0 += 32) {
        Frag16 a, b0, b1;
        a.h[0]  = *(const v8h*)(wcb + c0 + lh * 16);
        a.h[1]  = *(const v8h*)(wcb + c0 + lh * 16 + 8);
        b0.h[0] = *(const v8h*)(hb0 + c0 + lh * 16);
        b0.h[1] = *(const v8h*)(hb0 + c0 + lh * 16 + 8);
        b1.h[0] = *(const v8h*)(hb1 + c0 + lh * 16);
        b1.h[1] = *(const v8h*)(hb1 + c0 + lh * 16 + 8);
        acc0 = wmma_f16(a, b0, acc0);
        acc1 = wmma_f16(a, b1, acc1);
    }

    // ---- masked 3x3 conv: 5 live taps x 4 K-chunks, A reused for 2 N-tiles --
    const _Float16* wlb = WlF + (layer * HID + wave * 16 + lr) * 5 * HID;
    __builtin_prefetch(wlb, 0, 3);
    #pragma unroll
    for (int t = 0; t < 5; ++t) {
        const int ky = (t < 3) ? 0 : 1;          // halo row
        const int kx = (t < 3) ? t : (t - 3);
        const _Float16* sb0 = s_feat + (ky * HALOC + lr + kx) * HID;
        const _Float16* sb1 = sb0 + 16 * HID;
        const _Float16* wa  = wlb + t * HID;
        #pragma unroll
        for (int c0 = 0; c0 < HID; c0 += 32) {
            Frag16 a, b0, b1;
            a.h[0]  = *(const v8h*)(wa + c0 + lh * 16);
            a.h[1]  = *(const v8h*)(wa + c0 + lh * 16 + 8);
            b0.h[0] = *(const v8h*)(sb0 + c0 + lh * 16);
            b0.h[1] = *(const v8h*)(sb0 + c0 + lh * 16 + 8);
            b1.h[0] = *(const v8h*)(sb1 + c0 + lh * 16);
            b1.h[1] = *(const v8h*)(sb1 + c0 + lh * 16 + 8);
            acc0 = wmma_f16(a, b0, acc0);
            acc1 = wmma_f16(a, b1, acc1);
        }
    }

    // ---- bias + ReLU, D-fragment -> LDS transpose, coalesced f16 store ----
    #pragma unroll
    for (int v = 0; v < 8; ++v) {
        int o = wave * 16 + v + 8 * lh;
        float bias = Bl[layer * HID + o];
        float r0 = acc0[v] + bias;
        float r1 = acc1[v] + bias;
        s_out[lr * HID + o]        = (_Float16)(r0 > 0.f ? r0 : 0.f);
        s_out[(16 + lr) * HID + o] = (_Float16)(r1 > 0.f ? r1 : 0.f);
    }
    __syncthreads();
    #pragma unroll
    for (int it = 0; it < 2; ++it) {
        int idx = it * 256 + tid;        // 512 chunks of 16B
        int px  = idx >> 4;
        int c8  = idx & 15;
        v8h val = *(const v8h*)(s_out + px * HID + c8 * 8);
        *(v8h*)(actout + (((b * 32 + y) * 32 + px) * HID + c8 * 8)) = val;
    }
}

// ---------------------------------------------------------------------------
// Output GEMM: logits[768, p] = w_out @ act + b_out, fp32 out.
// Wave = 16(M) x 32(N), A-fragment reuse; act already post-ReLU.
// ---------------------------------------------------------------------------
__global__ __launch_bounds__(256, 2) void pixelcnn_out_kernel(
    const _Float16* __restrict__ act,    // [32768][128] post-relu
    const _Float16* __restrict__ WoF,    // [768][128]
    const float* __restrict__ bout,      // [768]
    float* __restrict__ out)             // [32][768][1024]
{
    const int tid  = threadIdx.x;
    const int blk  = blockIdx.x;
    const int nt   = blk & 1023;        // 32-pixel tile
    const int mb   = blk >> 10;         // 0..5 -> 128-row M block
    const int wave = tid >> 5;
    const int lane = tid & 31;
    const int lr   = lane & 15;
    const int lh   = lane >> 4;

    const int p0 = nt * 32 + lr;
    const _Float16* fb0 = act + p0 * HID;
    const _Float16* fb1 = fb0 + 16 * HID;
    const _Float16* wb  = WoF + (mb * HID + wave * 16 + lr) * HID;
    __builtin_prefetch(wb, 0, 3);

    v8f acc0 = {};
    v8f acc1 = {};
    #pragma unroll
    for (int c0 = 0; c0 < HID; c0 += 32) {
        Frag16 a, b0, b1;
        a.h[0]  = *(const v8h*)(wb + c0 + lh * 16);
        a.h[1]  = *(const v8h*)(wb + c0 + lh * 16 + 8);
        b0.h[0] = *(const v8h*)(fb0 + c0 + lh * 16);
        b0.h[1] = *(const v8h*)(fb0 + c0 + lh * 16 + 8);
        b1.h[0] = *(const v8h*)(fb1 + c0 + lh * 16);
        b1.h[1] = *(const v8h*)(fb1 + c0 + lh * 16 + 8);
        acc0 = wmma_f16(a, b0, acc0);
        acc1 = wmma_f16(a, b1, acc1);
    }

    const int bimg = p0 >> 10;          // both subtiles in same image/row
    const int pp   = p0 & 1023;
    #pragma unroll
    for (int v = 0; v < 8; ++v) {
        int o = mb * HID + wave * 16 + v + 8 * lh;
        float bias = bout[o];
        float* op = out + (bimg * 768 + o) * NPIX + pp;
        op[0]  = acc0[v] + bias;
        op[16] = acc1[v] + bias;
    }
}

// ---------------------------------------------------------------------------
extern "C" void kernel_launch(void* const* d_in, const int* in_sizes, int n_in,
                              void* d_out, int out_size, void* d_ws, size_t ws_size,
                              hipStream_t stream)
{
    const float* x    = (const float*)d_in[0];
    const float* h    = (const float*)d_in[1];
    const float* w0   = (const float*)d_in[2];
    const float* b0   = (const float*)d_in[3];
    const float* Wl   = (const float*)d_in[4];
    const float* Bl   = (const float*)d_in[5];
    const float* Wc   = (const float*)d_in[6];
    const float* wout = (const float*)d_in[7];
    const float* bout = (const float*)d_in[8];

    char* ws = (char*)d_ws;
    _Float16* featA = (_Float16*)(ws);                          // 8 MB
    _Float16* featB = (_Float16*)(ws + 8388608);                // 8 MB
    _Float16* hF    = (_Float16*)(ws + 16777216);               // 4 MB
    _Float16* WlF   = (_Float16*)(ws + 20971520);               // 983040 B
    _Float16* WcF   = (_Float16*)(ws + 20971520 + 983040);      // 98304 B
    _Float16* WoF   = (_Float16*)(ws + 20971520 + 983040 + 98304); // 196608 B

    pixelcnn_pack_kernel<<<8192, 256, 0, stream>>>(Wl, Wc, wout, h,
                                                   WlF, WcF, WoF, hF);
    pixelcnn_conv0_kernel<<<16384, 256, 0, stream>>>(x, w0, b0, featA);

    _Float16* fin = featA;
    _Float16* fout = featB;
    for (int l = 0; l < 6; ++l) {
        pixelcnn_layer_kernel<<<1024, 256, 0, stream>>>(fin, fout, WlF, WcF, hF,
                                                        Bl, l);
        _Float16* t = fin; fin = fout; fout = t;
    }
    pixelcnn_out_kernel<<<6144, 256, 0, stream>>>(fin, WoF, bout, (float*)d_out);
}
